// MiTA_Attention_15805479649452
// MI455X (gfx1250) — matrix operand order repaired and checked by reading, not compiled, not run on hardware
//
#include <hip/hip_runtime.h>
#include <math.h>

typedef __attribute__((ext_vector_type(16))) _Float16 v16h;
typedef __attribute__((ext_vector_type(2)))  _Float16 v2h;
typedef __attribute__((ext_vector_type(8)))  float    v8f;

namespace {
constexpr int kB    = 8;
constexpr int kGrid = 32;
constexpr int kN    = kGrid * kGrid + 1;   // 1025
constexpr int kC    = 1024;
constexpr int kC3   = 3 * kC;
constexpr int kH    = 16;
constexpr int kD    = 64;
constexpr int kPool = 5;
constexpr int kM    = 25;                  // routers / experts
constexpr int kTopK = 25;                  // selected keys per expert
constexpr int kBN   = kB * kN;             // 8200
constexpr float kScale = 0.125f;           // d^-0.5
}

// ---------- wave32 helpers (ds_bpermute based, no wave64 assumptions) ----------
__device__ __forceinline__ float bperm_f(float v, int srcLane) {
  return __int_as_float(__builtin_amdgcn_ds_bpermute(srcLane << 2, __float_as_int(v)));
}
__device__ __forceinline__ int bperm_i(int v, int srcLane) {
  return __builtin_amdgcn_ds_bpermute(srcLane << 2, v);
}
__device__ __forceinline__ float waveMax(float v, int lane) {
#pragma unroll
  for (int o = 16; o > 0; o >>= 1) v = fmaxf(v, bperm_f(v, lane ^ o));
  return v;
}
__device__ __forceinline__ float waveSum(float v, int lane) {
#pragma unroll
  for (int o = 16; o > 0; o >>= 1) v += bperm_f(v, lane ^ o);
  return v;
}

// ---------- 1) f32 -> f16 convert ----------
__global__ void __launch_bounds__(256) cvt_f16_kernel(const float* __restrict__ src,
                                                      _Float16* __restrict__ dst, int n) {
  int i = blockIdx.x * blockDim.x + threadIdx.x;
  if (i < n) dst[i] = (_Float16)src[i];
}

// ---------- 2) pack weight [K,N] row-major -> WMMA B-fragment order ----------
// frag index = (tn*KT + kt); within frag: lane (0..31), half h (0..15)
// element = W[kt*32 + (lane>=16?16:0) + h, tn*16 + (lane&15)]
__global__ void __launch_bounds__(256) pack_w_kernel(const float* __restrict__ W,
                                                     _Float16* __restrict__ Bp,
                                                     int KK, int NN) {
  int i = blockIdx.x * blockDim.x + threadIdx.x;
  if (i >= KK * NN) return;
  int frag = i >> 9, rem = i & 511;
  int lane = rem >> 4, h = rem & 15;
  int KT = KK >> 5;
  int kt = frag % KT, tn = frag / KT;
  int kk = kt * 32 + ((lane & 16) ? 16 : 0) + h;
  int nn = tn * 16 + (lane & 15);
  Bp[i] = (_Float16)W[(size_t)kk * NN + nn];
}

// ---------- 3) register-blocked f16 WMMA GEMM: 64x32 output per wave ----------
// Cout[MM,NN] = A[MM,KK] * Bp(packed) (+bias). k-loop unrolled x2 so the
// scheduler can keep ~24 b128 loads in flight behind 16 wmma per block.
__global__ void __launch_bounds__(128) gemm_f16_wmma_kernel(
    const _Float16* __restrict__ A, const _Float16* __restrict__ Bp,
    const float* __restrict__ bias, float* __restrict__ Cout,
    int MM, int NN, int KK) {
  constexpr int SM = 4, SN = 2;              // 4x2 sub-tiles of 16x16
  int wave   = blockIdx.x * (blockDim.x >> 5) + (threadIdx.x >> 5);
  int macroN = (NN >> 4) / SN;               // NN multiple of 32
  int macroM = (MM + SM * 16 - 1) / (SM * 16);
  if (wave >= macroM * macroN) return;
  int tm0 = (wave / macroN) * SM;
  int tn0 = (wave % macroN) * SN;
  int lane = threadIdx.x & 31;
  int hi   = lane >> 4;
  int KT   = KK >> 5;

  const _Float16* arow[SM];
#pragma unroll
  for (int i = 0; i < SM; ++i) {
    int row = (tm0 + i) * 16 + (lane & 15);
    int rc  = row < MM ? row : MM - 1;
    arow[i] = A + (size_t)rc * KK + hi * 16;
  }
  const v16h* bfrag[SN];
#pragma unroll
  for (int j = 0; j < SN; ++j)
    bfrag[j] = (const v16h*)Bp + ((size_t)(tn0 + j) * KT) * 32 + lane;

  v8f acc[SM][SN] = {};

#pragma unroll 2
  for (int kt = 0; kt < KT; ++kt) {
    v16h a[SM], b[SN];
#pragma unroll
    for (int i = 0; i < SM; ++i) {
      // A k order used here (contiguous, same as packed B): k = kt*32 + v*2 + hi*16 (+pos)
      const _Float16* ak = arow[i] + kt * 32;
#pragma unroll
      for (int v = 0; v < 8; ++v) {
        v2h p = *(const v2h*)(ak + v * 2);
        a[i][2 * v]     = p[0];
        a[i][2 * v + 1] = p[1];
      }
    }
#pragma unroll
    for (int j = 0; j < SN; ++j) b[j] = bfrag[j][(size_t)kt * 32];
#pragma unroll
    for (int i = 0; i < SM; ++i)
#pragma unroll
      for (int j = 0; j < SN; ++j)
        acc[i][j] = __builtin_amdgcn_wmma_f32_16x16x32_f16(
            false, a[i], false, b[j], (short)0, acc[i][j], false, false);
  }

#pragma unroll
  for (int i = 0; i < SM; ++i) {
#pragma unroll
    for (int j = 0; j < SN; ++j) {
      int   ncol = (tn0 + j) * 16 + (lane & 15);
      float bv   = bias ? bias[ncol] : 0.0f;
#pragma unroll
      for (int r = 0; r < 8; ++r) {
        int mrow = (tm0 + i) * 16 + hi * 8 + r;  // D layout: lanes16-31 hold M=8..15
        if (mrow < MM) Cout[(size_t)mrow * NN + ncol] = acc[i][j][r] + bv;
      }
    }
  }
}

// ---------- 4) split qkv -> per-head f16 q,k and f32 v ----------
__global__ void __launch_bounds__(256) split_qkv_kernel(
    const float* __restrict__ qkv, _Float16* __restrict__ hq,
    _Float16* __restrict__ hk, float* __restrict__ vf) {
  int i = blockIdx.x * blockDim.x + threadIdx.x;
  if (i >= kBN * kC) return;
  int c = i % kC; int bn = i / kC;
  int n = bn % kN; int b = bn / kN;
  int h = c >> 6, dd = c & 63;
  size_t src = (size_t)bn * kC3;
  size_t dst = (((size_t)(b * kH + h)) * kN + n) * kD + dd;
  hq[dst] = (_Float16)qkv[src + c];
  hk[dst] = (_Float16)qkv[src + kC + c];
  vf[dst] = qkv[src + 2 * kC + c];
}

// ---------- 5) adaptive avg-pool of q grid -> router tokens (per-head f16) ----------
__global__ void __launch_bounds__(256) router_pool_kernel(
    const float* __restrict__ qkv, _Float16* __restrict__ hr) {
  int i = blockIdx.x * blockDim.x + threadIdx.x;
  if (i >= kB * kM * kC) return;
  int c = i % kC; int t = i / kC;
  int m = t % kM; int b = t / kM;
  int pi = m / kPool, pj = m % kPool;
  int rs = (pi * kGrid) / kPool, re = ((pi + 1) * kGrid + kPool - 1) / kPool;
  int cs = (pj * kGrid) / kPool, ce = ((pj + 1) * kGrid + kPool - 1) / kPool;
  float s = 0.f;
  for (int r = rs; r < re; ++r)
    for (int cc = cs; cc < ce; ++cc)
      s += qkv[((size_t)b * kN + r * kGrid + cc) * kC3 + c];   // q part, no cls token
  s /= (float)((re - rs) * (ce - cs));
  int h = c >> 6, dd = c & 63;
  hr[(((size_t)(b * kH + h)) * kM + m) * kD + dd] = (_Float16)s;
}

// ---------- 6) r_logits = R*K^T, gate = R*Q^T via WMMA (K-transposed B loads are
//              contiguous 32B row segments because head dim = 64) ----------
__global__ void __launch_bounds__(128) router_logits_kernel(
    const _Float16* __restrict__ hr, const _Float16* __restrict__ hk,
    const _Float16* __restrict__ hq, float* __restrict__ rl, float* __restrict__ gate) {
  int bh = blockIdx.x;
  const _Float16* R  = hr + (size_t)bh * kM * kD;
  const _Float16* Kh = hk + (size_t)bh * kN * kD;
  const _Float16* Qh = hq + (size_t)bh * kN * kD;
  float* RL = rl   + (size_t)bh * kM * kN;
  float* GA = gate + (size_t)bh * kM * kN;
  int wid = threadIdx.x >> 5, lane = threadIdx.x & 31;
  int hi  = lane >> 4;
  int tilesN = (kN + 15) >> 4;   // 65
  for (int item = wid; item < 2 * tilesN; item += (blockDim.x >> 5)) {
    int mt = item / tilesN, nt = item % tilesN;
    int mr = mt * 16 + (lane & 15); if (mr >= kM) mr = 0;
    int nc = nt * 16 + (lane & 15); if (nc >= kN) nc = 0;
    const _Float16* ar = R + (size_t)mr * kD + hi * 16;
    v8f ck = {}, cq = {};
#pragma unroll
    for (int kt = 0; kt < 2; ++kt) {
      v16h a;
#pragma unroll
      for (int v = 0; v < 8; ++v) {
        v2h p = *(const v2h*)(ar + kt * 32 + v * 2);
        a[2 * v]     = p[0];
        a[2 * v + 1] = p[1];
      }
      int koff = kt * 32 + hi * 16;   // B-frag: lane n, halves = contiguous k range
      v16h bk = *(const v16h*)(Kh + (size_t)nc * kD + koff);
      v16h bq = *(const v16h*)(Qh + (size_t)nc * kD + koff);
      ck = __builtin_amdgcn_wmma_f32_16x16x32_f16(false, a, false, bk, (short)0, ck, false, false);
      cq = __builtin_amdgcn_wmma_f32_16x16x32_f16(false, a, false, bq, (short)0, cq, false, false);
    }
#pragma unroll
    for (int r = 0; r < 8; ++r) {
      int mm = mt * 16 + hi * 8 + r;
      int nn = nt * 16 + (lane & 15);
      if (mm < kM && nn < kN) {
        RL[(size_t)mm * kN + nn] = ck[r];
        GA[(size_t)mm * kN + nn] = cq[r];
      }
    }
  }
}

// ---------- 7) agent value: softmax(rl*scale) @ V per (b,h,m), one wave each ----------
__global__ void __launch_bounds__(128) agent_value_kernel(
    const float* __restrict__ rl, const float* __restrict__ vf, float* __restrict__ av) {
  int gw = blockIdx.x * (blockDim.x >> 5) + (threadIdx.x >> 5);
  if (gw >= kB * kH * kM) return;
  int bh = gw / kM;
  int lane = threadIdx.x & 31;
  const float* L = rl + (size_t)gw * kN;
  const float* V = vf + (size_t)bh * kN * kD;
  float mx = -3.0e38f;
  for (int n = lane; n < kN; n += 32) mx = fmaxf(mx, L[n] * kScale);
  mx = waveMax(mx, lane);
  float se = 0.f;
  for (int n = lane; n < kN; n += 32) se += __expf(L[n] * kScale - mx);
  se = waveSum(se, lane);
  float inv = 1.f / se;
  float a0 = 0.f, a1 = 0.f;
  for (int n = 0; n < kN; ++n) {
    float w = __expf(L[n] * kScale - mx) * inv;  // uniform broadcast load per wave
    a0 += w * V[(size_t)n * kD + lane];
    a1 += w * V[(size_t)n * kD + 32 + lane];
  }
  av[(size_t)gw * kD + lane]      = a0;
  av[(size_t)gw * kD + 32 + lane] = a1;
}

// ---------- 8) wave-parallel top-25: row cached in registers, 25 rounds of
//              lane-local argmax + bpermute butterfly argmax (low-index ties) ----------
__global__ void __launch_bounds__(128) topk_kernel(const float* __restrict__ rl,
                                                   int* __restrict__ tk) {
  constexpr int SLOTS = (kN + 31) / 32;   // 33
  int gw = blockIdx.x * (blockDim.x >> 5) + (threadIdx.x >> 5);
  if (gw >= kB * kH * kM) return;
  int lane = threadIdx.x & 31;
  const float* L = rl + (size_t)gw * kN;
  float vals[SLOTS];
#pragma unroll
  for (int s = 0; s < SLOTS; ++s) {
    int n = lane + 32 * s;
    vals[s] = (n < kN) ? L[n] : -3.0e38f;
  }
  int* T = tk + (size_t)gw * kTopK;
  for (int t = 0; t < kTopK; ++t) {
    float bv = -3.0e38f; int bs = 0;
#pragma unroll
    for (int s = 0; s < SLOTS; ++s)
      if (vals[s] > bv) { bv = vals[s]; bs = s; }
    int bi = lane + 32 * bs;
#pragma unroll
    for (int o = 16; o > 0; o >>= 1) {
      float ov = bperm_f(bv, lane ^ o);
      int   oi = bperm_i(bi, lane ^ o);
      if (ov > bv || (ov == bv && oi < bi)) { bv = ov; bi = oi; }
    }
    if (lane == 0) T[t] = bi;
    // invalidate winner (fully unrolled so vals[] stays in registers)
    bool mine = (bi & 31) == lane;
    int  ms   = bi >> 5;
#pragma unroll
    for (int s = 0; s < SLOTS; ++s)
      if (mine && s == ms) vals[s] = -3.0e38f;
  }
}

// ---------- 9) expert routing: argmax over m of gate[:,n] ----------
__global__ void __launch_bounds__(256) expert_kernel(const float* __restrict__ gate,
                                                     int* __restrict__ eidx) {
  int i = blockIdx.x * blockDim.x + threadIdx.x;
  if (i >= kB * kH * kN) return;
  int n = i % kN; int bh = i / kN;
  const float* G = gate + (size_t)bh * kM * kN + n;
  float best = G[0]; int bi = 0;
  for (int m = 1; m < kM; ++m) {
    float g = G[(size_t)m * kN];
    if (g > best) { best = g; bi = m; }
  }
  eidx[i] = bi;
}

// ---------- 10) merged softmax over [25 agents | 25 selected keys]; one wave/query ----------
__global__ void __launch_bounds__(128) mixed_attn_kernel(
    const float* __restrict__ gate, const int* __restrict__ tk, const int* __restrict__ eidx,
    const _Float16* __restrict__ hq, const _Float16* __restrict__ hk,
    const float* __restrict__ vf, const float* __restrict__ av,
    _Float16* __restrict__ hattn) {
  int gw = blockIdx.x * (blockDim.x >> 5) + (threadIdx.x >> 5);
  if (gw >= kB * kH * kN) return;
  int n = gw % kN; int bh = gw / kN;
  int b = bh >> 4, h = bh & 15;
  int lane = threadIdx.x & 31;
  int e = eidx[gw];
  int sel = 0;
  float la = -3.0e38f, lm = -3.0e38f;
  if (lane < kM) {
    sel = tk[((size_t)bh * kM + e) * kTopK + lane];
    la  = gate[((size_t)bh * kM + lane) * kN + n] * kScale;
    const _Float16* qr = hq + ((size_t)bh * kN + n)   * kD;
    const _Float16* kr = hk + ((size_t)bh * kN + sel) * kD;
    float dot = 0.f;
#pragma unroll
    for (int dd = 0; dd < kD; ++dd) dot += (float)qr[dd] * (float)kr[dd];
    lm = dot * kScale;
  }
  float mx  = waveMax(fmaxf(la, lm), lane);
  float ea  = (lane < kM) ? __expf(la - mx) : 0.f;
  float em  = (lane < kM) ? __expf(lm - mx) : 0.f;
  float inv = 1.f / waveSum(ea + em, lane);
  const float* AV = av + (size_t)bh * kM * kD;
  const float* V  = vf + (size_t)bh * kN * kD;
  float o0 = 0.f, o1 = 0.f;
  for (int j = 0; j < kM; ++j) {
    float wa = bperm_f(ea, j) * inv;
    o0 += wa * AV[(size_t)j * kD + lane];
    o1 += wa * AV[(size_t)j * kD + 32 + lane];
    float wm   = bperm_f(em, j) * inv;
    int   rowj = bperm_i(sel, j);
    o0 += wm * V[(size_t)rowj * kD + lane];
    o1 += wm * V[(size_t)rowj * kD + 32 + lane];
  }
  size_t base = ((size_t)b * kN + n) * kC + (size_t)h * kD;   // [B,N,C] with c = h*64+dd
  hattn[base + lane]      = (_Float16)o0;
  hattn[base + 32 + lane] = (_Float16)o1;
}

// ---------------------------------------------------------------------------
extern "C" void kernel_launch(void* const* d_in, const int* in_sizes, int n_in,
                              void* d_out, int out_size, void* d_ws, size_t ws_size,
                              hipStream_t stream) {
  (void)in_sizes; (void)n_in; (void)out_size; (void)ws_size;
  const float* x     = (const float*)d_in[0];
  const float* Wqkv  = (const float*)d_in[1];
  const float* Wproj = (const float*)d_in[2];
  const float* bproj = (const float*)d_in[3];
  float* out = (float*)d_out;

  char* ws = (char*)d_ws;
  size_t off = 0;
  auto take = [&](size_t bytes) {
    size_t o = off;
    off = (off + bytes + 255) & ~(size_t)255;
    return o;
  };
  _Float16* h_x    = (_Float16*)(ws + take((size_t)kBN * kC * 2));
  _Float16* h_Wq   = (_Float16*)(ws + take((size_t)kC * kC3 * 2));
  _Float16* h_Wp   = (_Float16*)(ws + take((size_t)kC * kC * 2));
  float*    qkv    = (float*)   (ws + take((size_t)kBN * kC3 * 4));
  _Float16* h_q    = (_Float16*)(ws + take((size_t)kB * kH * kN * kD * 2));
  _Float16* h_k    = (_Float16*)(ws + take((size_t)kB * kH * kN * kD * 2));
  float*    v_f    = (float*)   (ws + take((size_t)kB * kH * kN * kD * 4));
  _Float16* h_r    = (_Float16*)(ws + take((size_t)kB * kH * kM * kD * 2));
  float*    rl     = (float*)   (ws + take((size_t)kB * kH * kM * kN * 4));
  float*    gate   = (float*)   (ws + take((size_t)kB * kH * kM * kN * 4));
  float*    av     = (float*)   (ws + take((size_t)kB * kH * kM * kD * 4));
  int*      tk     = (int*)     (ws + take((size_t)kB * kH * kM * kTopK * 4));
  int*      eidx   = (int*)     (ws + take((size_t)kB * kH * kN * 4));
  _Float16* h_attn = (_Float16*)(ws + take((size_t)kBN * kC * 2));

  // 1) convert inputs, pack weights into WMMA B-fragment order
  cvt_f16_kernel<<<(kBN * kC + 255) / 256, 256, 0, stream>>>(x, h_x, kBN * kC);
  pack_w_kernel<<<(kC * kC3 + 255) / 256, 256, 0, stream>>>(Wqkv, h_Wq, kC, kC3);
  pack_w_kernel<<<(kC * kC + 255) / 256, 256, 0, stream>>>(Wproj, h_Wp, kC, kC);

  // 2) qkv = x @ Wqkv   (8200 x 3072 x 1024, f16 WMMA, f32 accum, 64x32/wave)
  {
    int macroM = (kBN + 63) / 64;
    int macroN = (kC3 / 16) / 2;
    int waves  = macroM * macroN;
    gemm_f16_wmma_kernel<<<(waves + 3) / 4, 128, 0, stream>>>(h_x, h_Wq, nullptr, qkv,
                                                              kBN, kC3, kC);
  }

  // 3) per-head splits + router pooling
  split_qkv_kernel<<<(kBN * kC + 255) / 256, 256, 0, stream>>>(qkv, h_q, h_k, v_f);
  router_pool_kernel<<<(kB * kM * kC + 255) / 256, 256, 0, stream>>>(qkv, h_r);

  // 4) router logits (vs K) and gate (vs Q) via WMMA
  router_logits_kernel<<<kB * kH, 128, 0, stream>>>(h_r, h_k, h_q, rl, gate);

  // 5) agent branch, top-25 (wave-parallel, non-destructive), expert routing
  agent_value_kernel<<<(kB * kH * kM + 3) / 4, 128, 0, stream>>>(rl, v_f, av);
  topk_kernel<<<(kB * kH * kM + 3) / 4, 128, 0, stream>>>(rl, tk);
  expert_kernel<<<(kB * kH * kN + 255) / 256, 256, 0, stream>>>(gate, eidx);

  // 6) merged softmax + combine (one wave per query)
  mixed_attn_kernel<<<(kB * kH * kN + 3) / 4, 128, 0, stream>>>(gate, tk, eidx,
                                                                h_q, h_k, v_f, av, h_attn);

  // 7) out = attn @ Wproj + bproj  (8200 x 1024 x 1024, f16 WMMA)
  {
    int macroM = (kBN + 63) / 64;
    int macroN = (kC / 16) / 2;
    int waves  = macroM * macroN;
    gemm_f16_wmma_kernel<<<(waves + 3) / 4, 128, 0, stream>>>(h_attn, h_Wp, bproj, out,
                                                              kBN, kC, kC);
  }
}
